// DecorrelatedBatchNorm1d_29686813950531
// MI455X (gfx1250) — matrix operand order repaired and checked by reading, not compiled, not run on hardware
//
#include <hip/hip_runtime.h>

// ---------------------------------------------------------------------------
// DecorrelatedBatchNorm1d (ZCA whitening) for MI455X / gfx1250, wave32 + WMMA.
//   B = 65536, N = 512, f32 in/out. cov^(-1/2) via Newton-Schulz (GEMM-only).
//   Big GEMMs: split-bf16 (3xbf16 fp32-emulation) WMMA w/ f32 acc.
//   NS core: f32 WMMA 16x16x4. LDS tiles feature-major: fragment = 2 b128s.
//   Whitening GEMM stages via GLOBAL_LOAD_ASYNC_TO_LDS_B128 (double-buffered).
// ---------------------------------------------------------------------------

#define BATCH    65536
#define NF       512
#define CSPLIT   8                  // deterministic K-split for covariance
#define KSLICE   (BATCH / CSPLIT)   // 8192
#define NS_ITERS 16
#define EPSV     1e-3f
#define DENOM    65535.0f           // B - 1

#if defined(__has_builtin)
#if __has_builtin(__builtin_amdgcn_global_load_async_to_lds_b128) && \
    __has_builtin(__builtin_amdgcn_s_wait_asynccnt)
#define HAS_ASYNC 1
#endif
#endif
#ifndef HAS_ASYNC
#define HAS_ASYNC 0
#endif

typedef __bf16        bf16x16 __attribute__((ext_vector_type(16)));
typedef float         f32x8   __attribute__((ext_vector_type(8)));
typedef float         f32x4   __attribute__((ext_vector_type(4)));
typedef float         f32x2   __attribute__((ext_vector_type(2)));
typedef unsigned int  u32x4   __attribute__((ext_vector_type(4)));
typedef int           i32x4   __attribute__((ext_vector_type(4)));

union frag_u { bf16x16 v; u32x4 q[2]; };  // one bf16 fragment = 2 x 16B

__device__ __forceinline__ unsigned short f2bf(float f) {
  unsigned int u = __float_as_uint(f);
  u += 0x7FFFu + ((u >> 16) & 1u);          // round-to-nearest-even
  return (unsigned short)(u >> 16);
}
__device__ __forceinline__ float bf2f(unsigned short h) {
  return __uint_as_float((unsigned int)h << 16);
}

#if HAS_ASYNC
// 16B global -> LDS async copy (per-lane addresses), tracked by ASYNCcnt.
__device__ __forceinline__ void async_cp16(const void* g, void* l) {
  __builtin_amdgcn_global_load_async_to_lds_b128(
      (__attribute__((address_space(1))) i32x4*)g,
      (__attribute__((address_space(3))) i32x4*)l, 0, 0);
}
#endif

// ---------------- Phase 1: column mean (deterministic 2-level tree) --------
__global__ void k_colsum_partial(const float* __restrict__ x,
                                 float* __restrict__ part) {
  const int b = blockIdx.x;                 // 256 blocks, 256 rows each
  const int c = threadIdx.x * 2;
  float s0 = 0.f, s1 = 0.f;
  const float* p = x + (size_t)b * 256 * NF + c;
  for (int r = 0; r < 256; ++r) { s0 += p[0]; s1 += p[1]; p += NF; }
  part[(size_t)b * NF + c]     = s0;
  part[(size_t)b * NF + c + 1] = s1;
}

__global__ void k_mean_final(const float* __restrict__ part,
                             float* __restrict__ mean) {
  const int c = blockIdx.x * 256 + threadIdx.x;
  float s = 0.f;
  for (int p = 0; p < 256; ++p) s += part[(size_t)p * NF + c];
  mean[c] = s * (1.0f / BATCH);
}

// ---------------- Phase 2: center + split-bf16 (hi + lo) -------------------
__global__ void k_center_split(const float* __restrict__ x,
                               const float* __restrict__ mean,
                               unsigned short* __restrict__ xch,
                               unsigned short* __restrict__ xcl) {
  const size_t gid = (size_t)blockIdx.x * blockDim.x + threadIdx.x;
  const size_t e = gid * 2;
  const int c = (int)(e & (NF - 1));
  const float v0 = x[e]     - mean[c];
  const float v1 = x[e + 1] - mean[c + 1];
  const unsigned short h0 = f2bf(v0), h1 = f2bf(v1);
  const unsigned short l0 = f2bf(v0 - bf2f(h0));
  const unsigned short l1 = f2bf(v1 - bf2f(h1));
  *(unsigned int*)(xch + e) = (unsigned int)h0 | ((unsigned int)h1 << 16);
  *(unsigned int*)(xcl + e) = (unsigned int)l0 | ((unsigned int)l1 << 16);
}

// ---------------- Phase 3: covariance partials, split-bf16 WMMA ------------
// Symmetric C = Xc^T Xc: upper-triangular 64x64 tiles only (36 of 64),
// mirrored on store. grid (36, 1, CSPLIT). 3xbf16: AhBh + AhBl + AlBh.
__global__ __launch_bounds__(256) void k_cov_wmma(
    const unsigned short* __restrict__ xch, const unsigned short* __restrict__ xcl,
    float* __restrict__ covpart) {
  __shared__ __align__(16) unsigned short Fah[64 * 32];  // A hi [feat][k]
  __shared__ __align__(16) unsigned short Fal[64 * 32];  // A lo
  __shared__ __align__(16) unsigned short Fbh[64 * 32];  // B hi
  __shared__ __align__(16) unsigned short Fbl[64 * 32];  // B lo
  int rem = blockIdx.x, ti = 0;
  while (rem >= 8 - ti) { rem -= 8 - ti; ++ti; }
  const int tj = ti + rem;
  const int i0 = ti * 64, j0 = tj * 64;
  const size_t kbase = (size_t)blockIdx.z * KSLICE;

  const int tid = threadIdx.x, lane = tid & 31, wave = tid >> 5;
  const int rgrp = wave >> 1, cgrp = wave & 1;
  const bool hi = lane >= 16; const int l15 = lane & 15;
  const int ml  = (rgrp << 4) + l15;
  const int nl0 = (cgrp << 5) + l15, nl1 = nl0 + 16;
  const int kk = tid >> 3;              // staging: k row 0..31
  const int f8 = (tid & 7) << 3;        // staging: feature 0,8,...,56
  const int aq0 = hi ? 16 : 0, aq1 = hi ? 24 : 8;    // A-frag K offsets
  const int bq0 = hi ? 16 : 0;                       // B-frag K offset

  f32x8 acc0 = {}, acc1 = {};

  for (int k0 = 0; k0 < KSLICE; k0 += 32) {
    const size_t row = kbase + k0 + kk;
    const u32x4 qah = *(const u32x4*)(xch + row * NF + i0 + f8);
    const u32x4 qal = *(const u32x4*)(xcl + row * NF + i0 + f8);
    const u32x4 qbh = *(const u32x4*)(xch + row * NF + j0 + f8);
    const u32x4 qbl = *(const u32x4*)(xcl + row * NF + j0 + f8);
#pragma unroll
    for (int j = 0; j < 8; ++j) {       // transpose-on-store -> [feat][k]
      const int o = (f8 + j) * 32 + kk, w = j >> 1, sh = (j & 1) * 16;
      Fah[o] = (unsigned short)(qah[w] >> sh);
      Fal[o] = (unsigned short)(qal[w] >> sh);
      Fbh[o] = (unsigned short)(qbh[w] >> sh);
      Fbl[o] = (unsigned short)(qbl[w] >> sh);
    }
    __syncthreads();

    frag_u ah, al, b0h, b0l, b1h, b1l;  // each fragment: two contiguous b128s
    ah.q[0]  = *(const u32x4*)&Fah[ml * 32 + aq0];
    ah.q[1]  = *(const u32x4*)&Fah[ml * 32 + aq1];
    al.q[0]  = *(const u32x4*)&Fal[ml * 32 + aq0];
    al.q[1]  = *(const u32x4*)&Fal[ml * 32 + aq1];
    b0h.q[0] = *(const u32x4*)&Fbh[nl0 * 32 + bq0];
    b0h.q[1] = *(const u32x4*)&Fbh[nl0 * 32 + bq0 + 8];
    b0l.q[0] = *(const u32x4*)&Fbl[nl0 * 32 + bq0];
    b0l.q[1] = *(const u32x4*)&Fbl[nl0 * 32 + bq0 + 8];
    b1h.q[0] = *(const u32x4*)&Fbh[nl1 * 32 + bq0];
    b1h.q[1] = *(const u32x4*)&Fbh[nl1 * 32 + bq0 + 8];
    b1l.q[0] = *(const u32x4*)&Fbl[nl1 * 32 + bq0];
    b1l.q[1] = *(const u32x4*)&Fbl[nl1 * 32 + bq0 + 8];
    acc0 = __builtin_amdgcn_wmma_f32_16x16x32_bf16(false, ah.v, false, b0h.v,
                                                   (short)0, acc0, false, false);
    acc0 = __builtin_amdgcn_wmma_f32_16x16x32_bf16(false, ah.v, false, b0l.v,
                                                   (short)0, acc0, false, false);
    acc0 = __builtin_amdgcn_wmma_f32_16x16x32_bf16(false, al.v, false, b0h.v,
                                                   (short)0, acc0, false, false);
    acc1 = __builtin_amdgcn_wmma_f32_16x16x32_bf16(false, ah.v, false, b1h.v,
                                                   (short)0, acc1, false, false);
    acc1 = __builtin_amdgcn_wmma_f32_16x16x32_bf16(false, ah.v, false, b1l.v,
                                                   (short)0, acc1, false, false);
    acc1 = __builtin_amdgcn_wmma_f32_16x16x32_bf16(false, al.v, false, b1h.v,
                                                   (short)0, acc1, false, false);
    __syncthreads();
  }

  float* outp = covpart + (size_t)blockIdx.z * NF * NF;
  const int rbase = i0 + (rgrp << 4) + (hi ? 8 : 0);
  const int c0 = j0 + nl0, c1 = j0 + nl1;
#pragma unroll
  for (int v = 0; v < 8; ++v) {
    const int r = rbase + v;
    outp[(size_t)r * NF + c0] = acc0[v];
    outp[(size_t)r * NF + c1] = acc1[v];
    if (ti != tj) {                     // mirror lower triangle
      outp[(size_t)c0 * NF + r] = acc0[v];
      outp[(size_t)c1 * NF + r] = acc1[v];
    }
  }
}

__global__ void k_cov_reduce(const float* __restrict__ covpart,
                             float* __restrict__ cov) {
  const int i = blockIdx.x * 256 + threadIdx.x;
  float s = 0.f;
  for (int p = 0; p < CSPLIT; ++p) s += covpart[(size_t)p * NF * NF + i];
  s *= (1.0f / DENOM);
  if ((i >> 9) == (i & (NF - 1))) s += EPSV;
  cov[i] = s;
}

// ---------------- Phase 4: Frobenius norm + Newton-Schulz init -------------
__global__ void k_fro_part(const float* __restrict__ cov, float* __restrict__ part) {
  __shared__ float sm[256];
  const int t = threadIdx.x;
  const int base = blockIdx.x * 1024 + t * 4;
  float s = 0.f;
#pragma unroll
  for (int e = 0; e < 4; ++e) { const float v = cov[base + e]; s += v * v; }
  sm[t] = s; __syncthreads();
  for (int o = 128; o > 0; o >>= 1) { if (t < o) sm[t] += sm[t + o]; __syncthreads(); }
  if (t == 0) part[blockIdx.x] = sm[0];
}

__global__ void k_fro_final(const float* __restrict__ part, float* __restrict__ normv) {
  __shared__ float sm[256];
  const int t = threadIdx.x;
  sm[t] = part[t]; __syncthreads();
  for (int o = 128; o > 0; o >>= 1) { if (t < o) sm[t] += sm[t + o]; __syncthreads(); }
  if (t == 0) normv[0] = sqrtf(sm[0]);
}

__global__ void k_init_YZ(const float* __restrict__ cov,
                          const float* __restrict__ normv,
                          float* __restrict__ Y, float* __restrict__ Z) {
  const int i = blockIdx.x * 256 + threadIdx.x;
  const float inv = 1.0f / normv[0];
  Y[i] = cov[i] * inv;
  Z[i] = ((i >> 9) == (i & (NF - 1))) ? 1.0f : 0.0f;
}

// ---------------- Phase 5: 512^3 f32 GEMM  C = alpha*I + beta*(A@B) --------
__global__ __launch_bounds__(256) void k_gemm512(
    float* __restrict__ C, const float* __restrict__ A,
    const float* __restrict__ B, float alpha, float beta) {
  __shared__ __align__(16) float Am[64 * 16];   // [m][k]
  __shared__ __align__(16) float Bn[64 * 16];   // [n][k]
  const int tid = threadIdx.x, lane = tid & 31, wave = tid >> 5;
  const int rgrp = wave >> 1, cgrp = wave & 1;
  const int i0 = blockIdx.x * 64, j0 = blockIdx.y * 64;
  const bool hi = lane >= 16; const int l15 = lane & 15;
  const int ml = (rgrp << 4) + l15;
  const int nl0 = (cgrp << 5) + l15, nl1 = nl0 + 16;
  f32x8 acc0 = {}, acc1 = {};

  const int am_m = tid >> 2, am_k4 = (tid & 3) << 2;   // A: direct [m][k]
  const int bk = tid >> 4,   bn4 = (tid & 15) << 2;    // B: transpose-on-store

  for (int k0 = 0; k0 < NF; k0 += 16) {
    *(f32x4*)&Am[am_m * 16 + am_k4] =
        *(const f32x4*)(A + (size_t)(i0 + am_m) * NF + k0 + am_k4);
    const f32x4 qb = *(const f32x4*)(B + (size_t)(k0 + bk) * NF + j0 + bn4);
#pragma unroll
    for (int j = 0; j < 4; ++j) Bn[(bn4 + j) * 16 + bk] = qb[j];
    __syncthreads();
#pragma unroll
    for (int kkk = 0; kkk < 16; kkk += 4) {   // f32 frag: 2 contiguous floats
      const f32x2 a  = *(const f32x2*)&Am[ml * 16 + kkk + (hi ? 2 : 0)];
      const f32x2 b0 = *(const f32x2*)&Bn[nl0 * 16 + kkk + (hi ? 2 : 0)];
      const f32x2 b1 = *(const f32x2*)&Bn[nl1 * 16 + kkk + (hi ? 2 : 0)];
      acc0 = __builtin_amdgcn_wmma_f32_16x16x4_f32(false, a, false, b0,
                                                   (short)0, acc0, false, false);
      acc1 = __builtin_amdgcn_wmma_f32_16x16x4_f32(false, a, false, b1,
                                                   (short)0, acc1, false, false);
    }
    __syncthreads();
  }
  const int rbase = i0 + (rgrp << 4) + (hi ? 8 : 0);
#pragma unroll
  for (int v = 0; v < 8; ++v) {
    const int r = rbase + v, c0 = j0 + nl0, c1 = j0 + nl1;
    C[(size_t)r * NF + c0] = beta * acc0[v] + (r == c0 ? alpha : 0.0f);
    C[(size_t)r * NF + c1] = beta * acc1[v] + (r == c1 ? alpha : 0.0f);
  }
}

// Z (f32, row-major) -> zbT hi/lo (bf16, transposed [n][k]).
__global__ void k_tobf16t(const float* __restrict__ Z,
                          unsigned short* __restrict__ zbTh,
                          unsigned short* __restrict__ zbTl) {
  const int gid = blockIdx.x * 256 + threadIdx.x;     // 1024 blocks
  const int r = gid >> 9, c = gid & (NF - 1);
  const float z = Z[gid];
  const unsigned short h = f2bf(z);
  zbTh[(size_t)c * NF + r] = h;
  zbTl[(size_t)c * NF + r] = f2bf(z - bf2f(h));
}

// ---------------- Phase 6: whitening GEMM + fused epilogue -----------------
// out(65536x512) = centered @ (Z / sqrt(s)) * weight + bias, split-bf16.
// Block tile 64x128, 8 waves x (16x64). A = xc[m][k], B = zbT[n][k].
// Async global->LDS staging, double-buffered, pipelined with s_wait_asynccnt.
__global__ __launch_bounds__(256) void k_out_wmma(
    const unsigned short* __restrict__ xch, const unsigned short* __restrict__ xcl,
    const unsigned short* __restrict__ zbTh, const unsigned short* __restrict__ zbTl,
    const float* __restrict__ weight, const float* __restrict__ bias,
    const float* __restrict__ normv, float* __restrict__ out) {
#if HAS_ASYNC
#define NBUF 2
#else
#define NBUF 1
#endif
  __shared__ __align__(16) unsigned short Amh[NBUF][64 * 32];   // A hi [m][k]
  __shared__ __align__(16) unsigned short Aml[NBUF][64 * 32];   // A lo
  __shared__ __align__(16) unsigned short Bnh[NBUF][128 * 32];  // B hi [n][k]
  __shared__ __align__(16) unsigned short Bnl[NBUF][128 * 32];  // B lo
  const int tid = threadIdx.x, lane = tid & 31, wave = tid >> 5;
  const int rg = wave >> 1, cg = wave & 1;
  const size_t m0 = (size_t)blockIdx.x * 64;
  const int n0 = blockIdx.y * 128;
  const bool hi = lane >= 16; const int l15 = lane & 15;
  const int ml = (rg << 4) + l15;
  const int aq0 = hi ? 16 : 0, aq1 = hi ? 24 : 8, bq0 = hi ? 16 : 0;
  f32x8 acc[4] = {};

  // staging maps: 16B per op; A planes 1 op/thread, B planes 2 ops/thread
  const int am_m = tid >> 2, am_k8 = (tid & 3) << 3;
  const int b0n = tid >> 2,         b0k8 = (tid & 3) << 3;
  const int b1n = (tid + 256) >> 2, b1k8 = b0k8;

  auto compute = [&](int buf) {
    frag_u ah, al;
    ah.q[0] = *(const u32x4*)&Amh[buf][ml * 32 + aq0];
    ah.q[1] = *(const u32x4*)&Amh[buf][ml * 32 + aq1];
    al.q[0] = *(const u32x4*)&Aml[buf][ml * 32 + aq0];
    al.q[1] = *(const u32x4*)&Aml[buf][ml * 32 + aq1];
#pragma unroll
    for (int s = 0; s < 4; ++s) {
      const int nl = (cg << 6) + (s << 4) + l15;
      frag_u bh, bl;
      const unsigned short* bph = &Bnh[buf][nl * 32 + bq0];
      const unsigned short* bpl = &Bnl[buf][nl * 32 + bq0];
      bh.q[0] = *(const u32x4*)bph; bh.q[1] = *(const u32x4*)(bph + 8);
      bl.q[0] = *(const u32x4*)bpl; bl.q[1] = *(const u32x4*)(bpl + 8);
      acc[s] = __builtin_amdgcn_wmma_f32_16x16x32_bf16(false, ah.v, false, bh.v,
                                                       (short)0, acc[s], false, false);
      acc[s] = __builtin_amdgcn_wmma_f32_16x16x32_bf16(false, ah.v, false, bl.v,
                                                       (short)0, acc[s], false, false);
      acc[s] = __builtin_amdgcn_wmma_f32_16x16x32_bf16(false, al.v, false, bh.v,
                                                       (short)0, acc[s], false, false);
    }
  };

#if HAS_ASYNC
  auto issue = [&](int buf, int k0) {
    async_cp16(xch + (m0 + am_m) * NF + k0 + am_k8, &Amh[buf][am_m * 32 + am_k8]);
    async_cp16(xcl + (m0 + am_m) * NF + k0 + am_k8, &Aml[buf][am_m * 32 + am_k8]);
    async_cp16(zbTh + (size_t)(n0 + b0n) * NF + k0 + b0k8, &Bnh[buf][b0n * 32 + b0k8]);
    async_cp16(zbTl + (size_t)(n0 + b0n) * NF + k0 + b0k8, &Bnl[buf][b0n * 32 + b0k8]);
    async_cp16(zbTh + (size_t)(n0 + b1n) * NF + k0 + b1k8, &Bnh[buf][b1n * 32 + b1k8]);
    async_cp16(zbTl + (size_t)(n0 + b1n) * NF + k0 + b1k8, &Bnl[buf][b1n * 32 + b1k8]);
  };
  issue(0, 0);
  for (int t = 0; t < NF / 32; ++t) {
    const int cur = t & 1;
    if (t + 1 < NF / 32) {
      issue(1 - cur, (t + 1) * 32);                  // prefetch next tile
      __builtin_amdgcn_s_wait_asynccnt(6);           // tile t landed
    } else {
      __builtin_amdgcn_s_wait_asynccnt(0);
    }
    __syncthreads();
    compute(cur);
    __syncthreads();                                 // before next overwrite
  }
#else
  for (int t = 0; t < NF / 32; ++t) {
    const int k0 = t * 32;
    *(u32x4*)&Amh[0][am_m * 32 + am_k8] =
        *(const u32x4*)(xch + (m0 + am_m) * NF + k0 + am_k8);
    *(u32x4*)&Aml[0][am_m * 32 + am_k8] =
        *(const u32x4*)(xcl + (m0 + am_m) * NF + k0 + am_k8);
    *(u32x4*)&Bnh[0][b0n * 32 + b0k8] =
        *(const u32x4*)(zbTh + (size_t)(n0 + b0n) * NF + k0 + b0k8);
    *(u32x4*)&Bnl[0][b0n * 32 + b0k8] =
        *(const u32x4*)(zbTl + (size_t)(n0 + b0n) * NF + k0 + b0k8);
    *(u32x4*)&Bnh[0][b1n * 32 + b1k8] =
        *(const u32x4*)(zbTh + (size_t)(n0 + b1n) * NF + k0 + b1k8);
    *(u32x4*)&Bnl[0][b1n * 32 + b1k8] =
        *(const u32x4*)(zbTl + (size_t)(n0 + b1n) * NF + k0 + b1k8);
    __syncthreads();
    compute(0);
    __syncthreads();
  }
#endif

  const float inv = rsqrtf(normv[0]);               // A^{-1/2} = Z / sqrt(s)
  const size_t rbase = m0 + (rg << 4) + (hi ? 8 : 0);
#pragma unroll
  for (int s = 0; s < 4; ++s) {
    const int c = n0 + (cg << 6) + (s << 4) + l15;
    const float w = weight[c] * inv, bb = bias[c];
#pragma unroll
    for (int v = 0; v < 8; ++v)
      out[(rbase + v) * NF + c] = acc[s][v] * w + bb;
  }
}

// ---------------------------------------------------------------------------
extern "C" void kernel_launch(void* const* d_in, const int* in_sizes, int n_in,
                              void* d_out, int out_size, void* d_ws, size_t ws_size,
                              hipStream_t stream) {
  const float* x      = (const float*)d_in[0];
  const float* weight = (const float*)d_in[1];
  const float* bias   = (const float*)d_in[2];
  float* out = (float*)d_out;
  char* ws = (char*)d_ws;

  const size_t MB = (size_t)1 << 20;
  float*          mean     = (float*)(ws + 0);             //   2 KB
  float*          normv    = (float*)(ws + 2048);          //  <1 KB
  float*          fropart  = (float*)(ws + 4096);          //   1 KB
  float*          meanpart = (float*)(ws + 8192);          // 512 KB
  float*          cov      = (float*)(ws + 1 * MB);        //   1 MB
  float*          Y        = (float*)(ws + 2 * MB);        //   1 MB
  float*          Z        = (float*)(ws + 3 * MB);        //   1 MB
  float*          T        = (float*)(ws + 4 * MB);        //   1 MB
  float*          U        = (float*)(ws + 5 * MB);        //   1 MB
  unsigned short* zbTh     = (unsigned short*)(ws + 6 * MB);             // 512 KB
  unsigned short* zbTl     = (unsigned short*)(ws + 6 * MB + 512 * 1024); // 512 KB
  float*          covpart  = (float*)(ws + 7 * MB);        //   8 MB
  unsigned short* xch      = (unsigned short*)(ws + 15 * MB);  // 64 MB
  unsigned short* xcl      = (unsigned short*)(ws + 79 * MB);  // 64 MB

  k_colsum_partial<<<256, 256, 0, stream>>>(x, meanpart);
  k_mean_final<<<2, 256, 0, stream>>>(meanpart, mean);
  k_center_split<<<(BATCH * NF) / 512, 256, 0, stream>>>(x, mean, xch, xcl);

  k_cov_wmma<<<dim3(36, 1, CSPLIT), 256, 0, stream>>>(xch, xcl, covpart);
  k_cov_reduce<<<(NF * NF) / 256, 256, 0, stream>>>(covpart, cov);

  k_fro_part<<<256, 256, 0, stream>>>(cov, fropart);
  k_fro_final<<<1, 256, 0, stream>>>(fropart, normv);
  k_init_YZ<<<(NF * NF) / 256, 256, 0, stream>>>(cov, normv, Y, Z);

  float *pY = Y, *pZ = Z, *pT = T, *pU = U;
  for (int it = 0; it < NS_ITERS; ++it) {
    k_gemm512<<<dim3(8, 8), 256, 0, stream>>>(pT, pZ, pY, 3.0f, -1.0f);
    k_gemm512<<<dim3(8, 8), 256, 0, stream>>>(pU, pY, pT, 0.0f, 0.5f);
    { float* t = pY; pY = pU; pU = t; }
    k_gemm512<<<dim3(8, 8), 256, 0, stream>>>(pU, pT, pZ, 0.0f, 0.5f);
    { float* t = pZ; pZ = pU; pU = t; }
  }

  k_tobf16t<<<(NF * NF) / 256, 256, 0, stream>>>(pZ, zbTh, zbTl);
  k_out_wmma<<<dim3(BATCH / 64, NF / 128), 256, 0, stream>>>(
      xch, xcl, zbTh, zbTl, weight, bias, normv, out);
}
